// DiscreteLatentSDE_47802986005058
// MI455X (gfx1250) — matrix operand order repaired and checked by reading, not compiled, not run on hardware
//
#include <hip/hip_runtime.h>
#include <math.h>

// ---------------------------------------------------------------------------
// Problem constants
// ---------------------------------------------------------------------------
#define T_IN   256
#define T_OUT  320
#define BN     256
#define DN     64
#define HN     128
#define GN     384          // 3*H
#define CN     64
#define LNN    64
#define ON     64
#define SN     4
#define NSTEP  (T_OUT - 1)  // 319
#define ROWS_GRU 32         // rows per GRU workgroup (8 chunks x 2 dirs)
#define ROWS_SDE 64         // rows per SDE workgroup (16 WGs * 64 = 1024)
#define RT_SDE  (ROWS_SDE / 16)

typedef __attribute__((ext_vector_type(16))) _Float16 v16h;
typedef __attribute__((ext_vector_type(8)))  float    v8f;

#define WMMA(a, b, c) \
  __builtin_amdgcn_wmma_f32_16x16x32_f16(false, (a), false, (b), (short)0, (c), false, false)

__device__ __forceinline__ v8f vzero8() {
  v8f z;
#pragma unroll
  for (int i = 0; i < 8; ++i) z[i] = 0.0f;
  return z;
}

// A fragment: 16 (rows) x 32 (K) f16 tile, p -> element (m0, k0), row-major, ld halves.
__device__ __forceinline__ v16h frag_a(const _Float16* p, int ld, int lane) {
  int m = lane & 15, hi = (lane >> 4) & 1;
  const _Float16* r0 = p + (size_t)m * ld + 8 * hi;
  const _Float16* r1 = r0 + 16;
  v16h a;
#pragma unroll
  for (int i = 0; i < 8; ++i) { a[i] = r0[i]; a[i + 8] = r1[i]; }
  return a;
}

// B fragment: B(k,n) = W[n][k] with W row-major (Nout, K); w -> element (n0, k0).
__device__ __forceinline__ v16h frag_b(const _Float16* w, int ld, int lane) {
  int n = lane & 15, hi = (lane >> 4) & 1;
  const _Float16* r = w + (size_t)n * ld + 8 * hi;
  v16h b;
#pragma unroll
  for (int i = 0; i < 8; ++i) { b[i] = r[i]; b[i + 8] = r[i + 16]; }
  return b;
}

// ---------------------------------------------------------------------------
// Branchless fast-math activations
// ---------------------------------------------------------------------------
__device__ __forceinline__ float fast_rcp(float x) { return __builtin_amdgcn_rcpf(x); }
__device__ __forceinline__ float sigmoidf_(float x) { return fast_rcp(1.0f + __expf(-x)); }
__device__ __forceinline__ float softplusf_(float x) {
  float ax = __builtin_fabsf(x);
  return fmaxf(x, 0.0f) + __logf(1.0f + __expf(-ax));
}
__device__ __forceinline__ float tanhf_(float x) {
  float ax = __builtin_fabsf(x);
  float e = __expf(-2.0f * ax);
  float r = (1.0f - e) * fast_rcp(1.0f + e);
  return __builtin_copysignf(r, x);
}

// ---------------------------------------------------------------------------
// fp32 -> fp16 cast
// ---------------------------------------------------------------------------
__global__ void cast_f32_f16(const float* __restrict__ s, _Float16* __restrict__ d, int n) {
  int i = blockIdx.x * blockDim.x + threadIdx.x;
  if (i < n) d[i] = (_Float16)s[i];
}

__global__ void zero2_kernel(float* p) { p[0] = 0.0f; p[1] = 0.0f; }

// ---------------------------------------------------------------------------
// Parallel input-projection GEMM: xp[dir][row][g] = x[row][:] @ wih[dir][g][:]
// ---------------------------------------------------------------------------
template <int K>
__global__ void xp_gemm_kernel(const _Float16* __restrict__ xin,  // [T*B][K]
                               const _Float16* __restrict__ wih,  // [2][G][K]
                               _Float16* __restrict__ xp) {       // [2][T*B][G]
  const int lane = threadIdx.x & 31, wave = threadIdx.x >> 5;
  const int CT = GN / 16;  // 24
  int unit = blockIdx.x * 8 + wave;
  int dir = unit & 1;
  int ct = (unit >> 1) % CT;
  int rt = unit / (2 * CT);  // 0 .. T*B/16-1
  const _Float16* arow = xin + (size_t)rt * 16 * K;
  const _Float16* wrow = wih + ((size_t)dir * GN + ct * 16) * K;
  __builtin_prefetch(wrow, 0, 1);  // warm L2 (global_prefetch path)
  v8f acc = vzero8();
#pragma unroll
  for (int k0 = 0; k0 < K; k0 += 32)
    acc = WMMA(frag_a(arow + k0, K, lane), frag_b(wrow + k0, K, lane), acc);
  int col = lane & 15, hi = (lane >> 4) & 1;
  _Float16* orow = xp + (size_t)dir * T_IN * BN * GN + (size_t)rt * 16 * GN + ct * 16 + col;
#pragma unroll
  for (int r = 0; r < 8; ++r) orow[(size_t)(r + 8 * hi) * GN] = (_Float16)acc[r];
}

// ---------------------------------------------------------------------------
// Persistent GRU recurrence (h part only; xp precomputed).
// grid = (8 row-chunks, 2 dirs), 256 threads.  Whh staged in LDS; h state
// double-buffered in LDS.  Unit loop kept rolled to avoid VGPR spills.
// ---------------------------------------------------------------------------
__global__ void gru_rec_kernel(const _Float16* __restrict__ xp,   // [2][T][B][G]
                               const _Float16* __restrict__ whh,  // [2][G][H]
                               const float* __restrict__ bih,
                               const float* __restrict__ bhh,
                               _Float16* __restrict__ outh,  // [T][B][2H]
                               float* __restrict__ fin,      // [B][512]
                               int layer) {
  extern __shared__ char smem_raw[];
  _Float16* wlds = (_Float16*)smem_raw;        // G x H = 96 KB
  _Float16* h0 = wlds + GN * HN;               // 32 x 128
  _Float16* h1 = h0 + ROWS_GRU * HN;

  const int tid = threadIdx.x, lane = tid & 31, wave = tid >> 5;
  const int dir = blockIdx.y;
  const int bbase = blockIdx.x * ROWS_GRU;

  const _Float16* whhd = whh + (size_t)dir * GN * HN;
  const float* bihd = bih + dir * GN;
  const float* bhhd = bhh + dir * GN;

  // stage Whh into LDS once; zero h
  for (int i = tid; i < GN * HN; i += blockDim.x) wlds[i] = whhd[i];
  for (int i = tid; i < ROWS_GRU * HN; i += blockDim.x) h0[i] = (_Float16)0.0f;
  __syncthreads();

  for (int t = 0; t < T_IN; ++t) {
    _Float16* hcur = (t & 1) ? h1 : h0;
    _Float16* hnxt = (t & 1) ? h0 : h1;
    int in_idx, out_idx;
    if (dir == 0) { in_idx = (layer == 0) ? (T_IN - 1 - t) : t; out_idx = t; }
    else          { in_idx = (layer == 0) ? t : (T_IN - 1 - t); out_idx = T_IN - 1 - t; }

    const _Float16* xpd = xp + ((size_t)dir * T_IN * BN + (size_t)in_idx * BN + bbase) * GN;

#pragma unroll 1
    for (int u = wave; u < 2 * 8; u += 8) {
      int mt = u >> 3;  // row tile 0..1
      int ct = u & 7;   // gate-column triple 0..7
      int gr = ct * 16, gz = HN + ct * 16, gn = 2 * HN + ct * 16;

      v8f hr = vzero8(), hz = vzero8(), hn = vzero8();
      const _Float16* hrow = hcur + (size_t)mt * 16 * HN;
#pragma unroll
      for (int k0 = 0; k0 < HN; k0 += 32) {
        v16h a = frag_a(hrow + k0, HN, lane);
        hr = WMMA(a, frag_b(wlds + (size_t)gr * HN + k0, HN, lane), hr);
        hz = WMMA(a, frag_b(wlds + (size_t)gz * HN + k0, HN, lane), hz);
        hn = WMMA(a, frag_b(wlds + (size_t)gn * HN + k0, HN, lane), hn);
      }

      int col = lane & 15, hi = (lane >> 4) & 1;
      float br = bihd[gr + col], bz = bihd[gz + col], bn = bihd[gn + col];
      float cr = bhhd[gr + col], cz = bhhd[gz + col], cn = bhhd[gn + col];
#pragma unroll
      for (int r = 0; r < 8; ++r) {
        int row = mt * 16 + r + 8 * hi;  // local row 0..31
        const _Float16* xg = xpd + (size_t)row * GN;
        float rg = sigmoidf_((float)xg[gr + col] + br + hr[r] + cr);
        float zg = sigmoidf_((float)xg[gz + col] + bz + hz[r] + cz);
        float ng = tanhf_((float)xg[gn + col] + bn + rg * (hn[r] + cn));
        float hold = (float)hcur[row * HN + ct * 16 + col];
        float hnew = (1.0f - zg) * ng + zg * hold;
        hnxt[row * HN + ct * 16 + col] = (_Float16)hnew;
        outh[((size_t)out_idx * BN + bbase + row) * (2 * HN) + dir * HN + ct * 16 + col] =
            (_Float16)hnew;
        if (t == T_IN - 1)
          fin[(size_t)(bbase + row) * 512 + layer * 256 + dir * 128 + ct * 16 + col] = hnew;
      }
    }
    __syncthreads();
  }
}

// ---------------------------------------------------------------------------
// LDS GEMM helper for the SDE MLPs: O = act(A @ W^T + bias), 64 rows fixed.
// ACT: 0 = none, 1 = softplus, 2 = sigmoid.
// ---------------------------------------------------------------------------
template <int ACT, int K>
__device__ __forceinline__ void gemm_lds(const _Float16* A, int lda,
                                         _Float16* O, int N,
                                         const _Float16* __restrict__ W,
                                         const float* __restrict__ bias,
                                         int wave, int lane) {
  const int ctiles = N >> 4;
  const int units = RT_SDE * ctiles;
#pragma unroll 1
  for (int u = wave; u < units; u += 8) {
    int mt = u / ctiles, nt = u % ctiles;
    v8f acc = vzero8();
    const _Float16* arow = A + (size_t)mt * 16 * lda;
#pragma unroll
    for (int k0 = 0; k0 < K; k0 += 32) {
      acc = WMMA(frag_a(arow + k0, lda, lane),
                 frag_b(W + (size_t)(nt * 16) * K + k0, K, lane), acc);
    }
    int col = lane & 15, hi = (lane >> 4) & 1;
    float bv = bias[nt * 16 + col];
#pragma unroll
    for (int r = 0; r < 8; ++r) {
      float v = acc[r] + bv;
      if (ACT == 1) v = softplusf_(v);
      else if (ACT == 2) v = sigmoidf_(v);
      O[(mt * 16 + r + 8 * hi) * N + nt * 16 + col] = (_Float16)v;
    }
  }
}

// ---------------------------------------------------------------------------
// Persistent SDE rollout: 16 WGs x 64 rows, 319 steps, all state in LDS.
// ---------------------------------------------------------------------------
__global__ void sde_kernel(const float* __restrict__ zrow,    // [1024][64]
                           const float* __restrict__ ctxbuf,  // [256][B][64]
                           const float* __restrict__ noise,   // [319][B][S][L]
                           const _Float16* fw1, const float* fb1,
                           const _Float16* fw2, const float* fb2,
                           const _Float16* fw3, const float* fb3,
                           const _Float16* hw1, const float* hb1,
                           const _Float16* hw2, const float* hb2,
                           const _Float16* hw3, const float* hb3,
                           const _Float16* gw1, const float* gb1,
                           const _Float16* gw2, const float* gb2,
                           float* __restrict__ zsg,  // [320][B][64]
                           float* __restrict__ logqp_acc) {
  extern __shared__ char smem_raw[];
  float*    zf   = (float*)smem_raw;                  // 64 x 64 f32
  _Float16* catb = (_Float16*)(zf + ROWS_SDE * LNN);  // 64 x 128
  _Float16* u1 = catb + ROWS_SDE * 128;
  _Float16* v1 = u1 + ROWS_SDE * 128;
  _Float16* w1 = v1 + ROWS_SDE * 128;
  _Float16* u2 = w1 + ROWS_SDE * 128;
  _Float16* v2 = u2 + ROWS_SDE * 128;
  _Float16* drf = v2 + ROWS_SDE * 128;                // 64 x 64
  _Float16* hdr = drf + ROWS_SDE * LNN;
  _Float16* dif = hdr + ROWS_SDE * LNN;

  const int tid = threadIdx.x, lane = tid & 31, wave = tid >> 5;
  const int r0 = blockIdx.x * ROWS_SDE;
  const float alpha = 0.1f, sqa = 0.31622776601683794f;

  __builtin_prefetch(fw1, 0, 1);
  __builtin_prefetch(fw2, 0, 1);
  __builtin_prefetch(hw2, 0, 1);

  for (int i = tid; i < ROWS_SDE * LNN; i += blockDim.x) zf[i] = zrow[(size_t)r0 * LNN + i];
  __syncthreads();

  float klacc = 0.0f;
  for (int t = 0; t < NSTEP; ++t) {
    int idxt = (t + 1 < T_IN) ? (t + 1) : (T_IN - 1);
    // assemble [z | ctx]
    for (int i = tid; i < ROWS_SDE * 128; i += blockDim.x) {
      int rr = i >> 7, cc = i & 127;
      float v;
      if (cc < LNN) v = zf[rr * LNN + cc];
      else {
        int b = (r0 + rr) >> 2;
        v = ctxbuf[((size_t)idxt * BN + b) * CN + (cc - LNN)];
      }
      catb[i] = (_Float16)v;
    }
    __syncthreads();

    // stage 1: three independent first layers
    gemm_lds<1, 128>(catb, 128, u1, 128, fw1, fb1, wave, lane);
    gemm_lds<1, 64>(catb, 128, v1, 128, hw1, hb1, wave, lane);
    gemm_lds<1, 64>(catb, 128, w1, 128, gw1, gb1, wave, lane);
    __syncthreads();
    // stage 2
    gemm_lds<1, 128>(u1, 128, u2, 128, fw2, fb2, wave, lane);
    gemm_lds<1, 128>(v1, 128, v2, 128, hw2, hb2, wave, lane);
    gemm_lds<2, 128>(w1, 128, dif, 64, gw2, gb2, wave, lane);
    __syncthreads();
    // stage 3
    gemm_lds<0, 128>(u2, 128, drf, 64, fw3, fb3, wave, lane);
    gemm_lds<0, 128>(v2, 128, hdr, 64, hw3, hb3, wave, lane);
    __syncthreads();

    // Euler-Maruyama + KL
    for (int i = tid; i < ROWS_SDE * LNN; i += blockDim.x) {
      int ll = i & 63;
      int grow = r0 + (i >> 6), b = grow >> 2, s = grow & 3;
      float z = zf[i];
      float pm = z + alpha * (float)drf[i];
      float ps = sqa * (float)dif[i];
      float hm = z + alpha * (float)hdr[i];
      float nz = noise[(((size_t)t * BN + b) * SN + s) * LNN + ll];
      float d = (pm - hm) / ps;
      klacc += 0.5f * d * d;
      zf[i] = pm + ps * nz;
    }
    __syncthreads();

    // zs[t+1] = mean over samples
    for (int i = tid; i < (ROWS_SDE / SN) * LNN; i += blockDim.x) {
      int bl = i >> 6, ll = i & 63;
      float m = 0.25f * (zf[(bl * 4 + 0) * LNN + ll] + zf[(bl * 4 + 1) * LNN + ll] +
                         zf[(bl * 4 + 2) * LNN + ll] + zf[(bl * 4 + 3) * LNN + ll]);
      zsg[((size_t)(t + 1) * BN + (r0 >> 2) + bl) * LNN + ll] = m;
    }
    __syncthreads();
  }

  float* red = (float*)catb;
  red[tid] = klacc;
  __syncthreads();
  for (int s = 128; s > 0; s >>= 1) {
    if (tid < s) red[tid] += red[tid + s];
    __syncthreads();
  }
  if (tid == 0) atomicAdd(logqp_acc, red[0] * (1.0f / (SN * BN)));
}

// ---------------------------------------------------------------------------
// Tails
// ---------------------------------------------------------------------------
__global__ void qz0_kernel(const float* __restrict__ fin, const float* __restrict__ w,
                           const float* __restrict__ b, float* qm, float* qls) {
  int j = blockIdx.x * blockDim.x + threadIdx.x;  // 256*128
  int bb = j >> 7, c = j & 127;
  float acc = b[c];
  const float* fr = fin + (size_t)bb * 512;
  const float* wr = w + (size_t)c * 512;
  for (int k = 0; k < 512; ++k) acc += fr[k] * wr[k];
  if (c < 64) qm[bb * 64 + c] = acc;
  else qls[bb * 64 + (c - 64)] = acc;
}

__global__ void ctx_kernel(const _Float16* __restrict__ out1, const float* __restrict__ w,
                           const float* __restrict__ b, float* __restrict__ ctxbuf) {
  int flat = blockIdx.x * blockDim.x + threadIdx.x;  // 256*256*64
  int c = flat & 63, bb = (flat >> 6) & 255, i = flat >> 14;
  const _Float16* xr = out1 + ((size_t)(T_IN - 1 - i) * BN + bb) * 256;
  const float* wr = w + (size_t)c * 256;
  float acc = b[c];
  for (int k = 0; k < 256; ++k) acc += (float)xr[k] * wr[k];
  ctxbuf[flat] = acc;
}

__global__ void z0_kernel(const float* qm, const float* qls, const float* nz, float* zrow) {
  int i = blockIdx.x * blockDim.x + threadIdx.x;  // B*S*L
  int l = i & 63, s = (i >> 6) & 3, bb = i >> 8;
  zrow[i] = qm[bb * 64 + l] + __expf(qls[bb * 64 + l]) * nz[((size_t)bb * 64 + l) * 4 + s];
}

__global__ void zs0_kernel(const float* zrow, float* zs) {
  int i = blockIdx.x * blockDim.x + threadIdx.x;  // 256*64
  int l = i & 63, bb = i >> 6;
  zs[i] = 0.25f * (zrow[(bb * 4 + 0) * 64 + l] + zrow[(bb * 4 + 1) * 64 + l] +
                   zrow[(bb * 4 + 2) * 64 + l] + zrow[(bb * 4 + 3) * 64 + l]);
}

__global__ void recon_kernel(const float* __restrict__ zs, const float* __restrict__ xs,
                             const float* __restrict__ muw, const float* __restrict__ mub,
                             const float* __restrict__ lsw, const float* __restrict__ lsb,
                             float* acc_out) {
  __shared__ float red[256];
  int flat = blockIdx.x * blockDim.x + threadIdx.x;  // T_IN*B*D
  int d = flat & 63, bb = (flat >> 6) & 255, t = flat >> 14;
  const float* zr = zs + ((size_t)t * BN + bb) * 64;
  float mu = mub[d], ls = lsb[d];
  for (int k = 0; k < 64; ++k) { float zv = zr[k]; mu += zv * muw[d * 64 + k]; ls += zv * lsw[d * 64 + k]; }
  ls = fminf(fmaxf(ls, -10.0f), 2.0f);
  float st = __expf(ls);
  float dd = (xs[flat] - mu) / st;
  red[threadIdx.x] = -0.5f * dd * dd - ls - 0.91893853320467274f;
  __syncthreads();
  for (int s = 128; s > 0; s >>= 1) {
    if (threadIdx.x < s) red[threadIdx.x] += red[threadIdx.x + s];
    __syncthreads();
  }
  if (threadIdx.x == 0) atomicAdd(acc_out, red[0] * (1.0f / 256.0f));
}

__global__ void outproj_kernel(const float* __restrict__ zs, const float* __restrict__ w,
                               const float* __restrict__ b, float* __restrict__ outp) {
  int flat = blockIdx.x * blockDim.x + threadIdx.x;  // 64*256*64
  int o = flat & 63, bb = (flat >> 6) & 255, tt = flat >> 14;
  const float* zr = zs + ((size_t)(T_IN + tt) * BN + bb) * 64;
  float acc = b[o];
  for (int k = 0; k < 64; ++k) acc += zr[k] * w[o * 64 + k];
  outp[flat] = acc;
}

__global__ void logqp0_kernel(const float* qm, const float* qls, const float* pm,
                              const float* pls, float* acc_out) {
  __shared__ float red[256];
  int i = blockIdx.x * blockDim.x + threadIdx.x;  // 256*64
  int l = i & 63;
  float ql = qls[i], qmv = qm[i], plv = pls[l], pmv = pm[l];
  float q2 = __expf(2.0f * ql), p2 = __expf(2.0f * plv);
  float dmu = qmv - pmv;
  red[threadIdx.x] = plv - ql + (q2 + dmu * dmu) / (2.0f * p2) - 0.5f;
  __syncthreads();
  for (int s = 128; s > 0; s >>= 1) {
    if (threadIdx.x < s) red[threadIdx.x] += red[threadIdx.x + s];
    __syncthreads();
  }
  if (threadIdx.x == 0) atomicAdd(acc_out, red[0] * (1.0f / 256.0f));
}

// ---------------------------------------------------------------------------
// Launcher
// ---------------------------------------------------------------------------
extern "C" void kernel_launch(void* const* d_in, const int* in_sizes, int n_in,
                              void* d_out, int out_size, void* d_ws, size_t ws_size,
                              hipStream_t stream) {
  (void)in_sizes; (void)n_in; (void)out_size; (void)ws_size;
  const float* xs        = (const float*)d_in[0];
  const float* z0_noise  = (const float*)d_in[3];
  const float* path_noise= (const float*)d_in[4];
  const float* gru0_wih  = (const float*)d_in[5];
  const float* gru0_whh  = (const float*)d_in[6];
  const float* gru0_bih  = (const float*)d_in[7];
  const float* gru0_bhh  = (const float*)d_in[8];
  const float* gru1_wih  = (const float*)d_in[9];
  const float* gru1_whh  = (const float*)d_in[10];
  const float* gru1_bih  = (const float*)d_in[11];
  const float* gru1_bhh  = (const float*)d_in[12];
  const float* qz0_w = (const float*)d_in[13]; const float* qz0_b = (const float*)d_in[14];
  const float* ctx_w = (const float*)d_in[15]; const float* ctx_b = (const float*)d_in[16];
  const float* f_w1 = (const float*)d_in[17]; const float* f_b1 = (const float*)d_in[18];
  const float* f_w2 = (const float*)d_in[19]; const float* f_b2 = (const float*)d_in[20];
  const float* f_w3 = (const float*)d_in[21]; const float* f_b3 = (const float*)d_in[22];
  const float* h_w1 = (const float*)d_in[23]; const float* h_b1 = (const float*)d_in[24];
  const float* h_w2 = (const float*)d_in[25]; const float* h_b2 = (const float*)d_in[26];
  const float* h_w3 = (const float*)d_in[27]; const float* h_b3 = (const float*)d_in[28];
  const float* g_w1 = (const float*)d_in[29]; const float* g_b1 = (const float*)d_in[30];
  const float* g_w2 = (const float*)d_in[31]; const float* g_b2 = (const float*)d_in[32];
  const float* mu_w = (const float*)d_in[33]; const float* mu_b = (const float*)d_in[34];
  const float* ls_w = (const float*)d_in[35]; const float* ls_b = (const float*)d_in[36];
  const float* out_w = (const float*)d_in[37]; const float* out_b = (const float*)d_in[38];
  const float* pz0_mean = (const float*)d_in[39]; const float* pz0_logstd = (const float*)d_in[40];

  char* ws = (char*)d_ws;
  size_t off = 0;
  auto alloc = [&](size_t bytes) -> char* {
    char* p = ws + off;
    off += (bytes + 255) & ~(size_t)255;
    return p;
  };

  _Float16* xh    = (_Float16*)alloc((size_t)T_IN * BN * DN * 2);
  _Float16* out0h = (_Float16*)alloc((size_t)T_IN * BN * 256 * 2);
  _Float16* out1h = (_Float16*)alloc((size_t)T_IN * BN * 256 * 2);
  _Float16* xpbuf = (_Float16*)alloc((size_t)2 * T_IN * BN * GN * 2);  // reused by both layers
  _Float16* g0wih_h = (_Float16*)alloc((size_t)2 * GN * 64 * 2);
  _Float16* g0whh_h = (_Float16*)alloc((size_t)2 * GN * 128 * 2);
  _Float16* g1wih_h = (_Float16*)alloc((size_t)2 * GN * 256 * 2);
  _Float16* g1whh_h = (_Float16*)alloc((size_t)2 * GN * 128 * 2);
  _Float16* fw1h = (_Float16*)alloc(128 * 128 * 2);
  _Float16* fw2h = (_Float16*)alloc(128 * 128 * 2);
  _Float16* fw3h = (_Float16*)alloc(64 * 128 * 2);
  _Float16* hw1h = (_Float16*)alloc(128 * 64 * 2);
  _Float16* hw2h = (_Float16*)alloc(128 * 128 * 2);
  _Float16* hw3h = (_Float16*)alloc(64 * 128 * 2);
  _Float16* gw1h = (_Float16*)alloc(128 * 64 * 2);
  _Float16* gw2h = (_Float16*)alloc(64 * 128 * 2);
  float* finb   = (float*)alloc((size_t)BN * 512 * 4);
  float* qz0m   = (float*)alloc((size_t)BN * 64 * 4);
  float* qz0ls  = (float*)alloc((size_t)BN * 64 * 4);
  float* ctxbuf = (float*)alloc((size_t)T_IN * BN * CN * 4);
  float* zrow   = (float*)alloc((size_t)BN * SN * LNN * 4);
  float* zsbuf  = (float*)alloc((size_t)T_OUT * BN * LNN * 4);

  float* outp = (float*)d_out;
  float* scal = outp + (size_t)(T_OUT - T_IN) * BN * ON;  // [log_pxs, logqp]

  auto cast = [&](const float* s, _Float16* d, int n) {
    cast_f32_f16<<<(n + 255) / 256, 256, 0, stream>>>(s, d, n);
  };

  zero2_kernel<<<1, 1, 0, stream>>>(scal);

  cast(xs, xh, T_IN * BN * DN);
  cast(gru0_wih, g0wih_h, 2 * GN * 64);
  cast(gru0_whh, g0whh_h, 2 * GN * 128);
  cast(gru1_wih, g1wih_h, 2 * GN * 256);
  cast(gru1_whh, g1whh_h, 2 * GN * 128);
  cast(f_w1, fw1h, 128 * 128); cast(f_w2, fw2h, 128 * 128); cast(f_w3, fw3h, 64 * 128);
  cast(h_w1, hw1h, 128 * 64);  cast(h_w2, hw2h, 128 * 128); cast(h_w3, hw3h, 64 * 128);
  cast(g_w1, gw1h, 128 * 64);  cast(g_w2, gw2h, 64 * 128);

  const int xp_tiles = (T_IN * BN / 16) * (GN / 16) * 2;  // 196608
  dim3 ggrid(BN / ROWS_GRU, 2);
  size_t gru_shmem = (size_t)GN * HN * sizeof(_Float16)               // Whh
                   + (size_t)2 * ROWS_GRU * HN * sizeof(_Float16);    // h double buffer

  // layer 0: parallel xp GEMM, then h-only recurrence
  xp_gemm_kernel<64><<<xp_tiles / 8, 256, 0, stream>>>(xh, g0wih_h, xpbuf);
  gru_rec_kernel<<<ggrid, 256, gru_shmem, stream>>>(
      xpbuf, g0whh_h, gru0_bih, gru0_bhh, out0h, finb, 0);
  // layer 1
  xp_gemm_kernel<256><<<xp_tiles / 8, 256, 0, stream>>>(out0h, g1wih_h, xpbuf);
  gru_rec_kernel<<<ggrid, 256, gru_shmem, stream>>>(
      xpbuf, g1whh_h, gru1_bih, gru1_bhh, out1h, finb, 1);

  qz0_kernel<<<(BN * 128) / 256, 256, 0, stream>>>(finb, qz0_w, qz0_b, qz0m, qz0ls);
  ctx_kernel<<<(T_IN * BN * CN) / 256, 256, 0, stream>>>(out1h, ctx_w, ctx_b, ctxbuf);
  z0_kernel<<<(BN * SN * LNN) / 256, 256, 0, stream>>>(qz0m, qz0ls, z0_noise, zrow);
  zs0_kernel<<<(BN * LNN) / 256, 256, 0, stream>>>(zrow, zsbuf);

  size_t sde_shmem = (size_t)ROWS_SDE * LNN * 4         // zf
                   + (size_t)ROWS_SDE * 128 * 2 * 6     // cat, u1, v1, w1, u2, v2
                   + (size_t)ROWS_SDE * LNN * 2 * 3;    // drf, hdr, dif
  sde_kernel<<<(BN * SN) / ROWS_SDE, 256, sde_shmem, stream>>>(
      zrow, ctxbuf, path_noise,
      fw1h, f_b1, fw2h, f_b2, fw3h, f_b3,
      hw1h, h_b1, hw2h, h_b2, hw3h, h_b3,
      gw1h, g_b1, gw2h, g_b2,
      zsbuf, scal + 1);

  recon_kernel<<<(T_IN * BN * DN) / 256, 256, 0, stream>>>(
      zsbuf, xs, mu_w, mu_b, ls_w, ls_b, scal);
  outproj_kernel<<<((T_OUT - T_IN) * BN * ON) / 256, 256, 0, stream>>>(
      zsbuf, out_w, out_b, outp);
  logqp0_kernel<<<(BN * LNN) / 256, 256, 0, stream>>>(
      qz0m, qz0ls, pz0_mean, pz0_logstd, scal + 1);
}
